// TCNLinearAttention_72043781423340
// MI455X (gfx1250) — matrix operand order repaired and checked by reading, not compiled, not run on hardware
//
#include <hip/hip_runtime.h>
#include <hip/hip_bf16.h>

typedef __bf16 bf16;
typedef __attribute__((ext_vector_type(16))) __bf16 v16bf;
typedef __attribute__((ext_vector_type(8)))  float   v8f;

#define B_  8
#define S_  4096
#define IN_ 64
#define D_  512
#define H_  8
#define HD_ 64
#define NL_ 2

// ---------------------------------------------------------------------------
// WMMA fragment helpers (layouts per CDNA5 ISA 7.12.2, wave32)
// A (16x32 bf16): lane L holds row M=L&15; K = ((L>>4)*8)+j (j<8) and 16+((L>>4)*8)+(j-8)
// B (32x16 bf16): lane L holds col N=L&15; K = ((L>>4)*8)+j (j<8) and 16+((L>>4)*8)+(j-8)   (from LDS transposed tile)
// C (16x16 f32):  lane L elem r -> M = 8*(L>>4)+r, N = L&15
// ---------------------------------------------------------------------------

#define LDT 40  // LDS row stride (bf16 elems), padded to dodge bank conflicts

__device__ __forceinline__ v16bf frag_a(const bf16* sm, int row, int lane) {
  v16bf f;
  const bf16* p = sm + row * LDT;
  int kb = (lane >> 4) * 8;
#pragma unroll
  for (int j = 0; j < 8; ++j) { f[j] = p[kb + j]; f[8 + j] = p[16 + kb + j]; }
  return f;
}
__device__ __forceinline__ v16bf frag_b(const bf16* sm, int col, int lane) {
  v16bf f;
  const bf16* p = sm + col * LDT;
  int kb = (lane >> 4) * 8;
#pragma unroll
  for (int j = 0; j < 8; ++j) { f[j] = p[kb + j]; f[8 + j] = p[16 + kb + j]; }
  return f;
}

// ---------------------------------------------------------------------------
// Generic batched GEMM:  C[z] = A[z](MxK,row) @ B[z](KxN,row) (+bias[N]) (+resid)
// Block = 128 thr (4 waves), 64x64 tile, each wave 32x32 via 2x2 WMMA.
// Epilogue options are compile-time to keep the hot path branch-free.
// Next-tile global_prefetch_b8 overlaps cacheline fill with WMMA work.
// ---------------------------------------------------------------------------
template <bool HAS_BIAS, bool HAS_RESID, bool HAS_OUTF, bool HAS_OUTB>
__global__ __launch_bounds__(128)
void k_gemm(const bf16* __restrict__ A, const bf16* __restrict__ Bm,
            const float* __restrict__ bias, const float* resid,
            float* outF, bf16* outB,
            int M, int N, int K,
            long long sA, long long sB, long long sR, long long sO) {
  __shared__ bf16 As[64 * LDT];
  __shared__ bf16 Bt[64 * LDT];
  const int tid = threadIdx.x, lane = tid & 31, wid = tid >> 5;
  const int wr = wid >> 1, wc = wid & 1;
  const int z = blockIdx.z;
  const int m0 = blockIdx.y * 64, n0 = blockIdx.x * 64;
  const bf16* Az = A + (long long)z * sA;
  const bf16* Bz = Bm + (long long)z * sB;

  v8f acc[2][2] = {};
  const int arow = tid >> 1, ahalf = (tid & 1) * 16;   // A: 64 rows x 2 halves
  const int bk = tid & 31, bn0 = (tid >> 5) * 16;      // B: 32 k-rows x 4 n-chunks

  for (int k0 = 0; k0 < K; k0 += 32) {
    __syncthreads();
    {   // A tile 64x32 (vectorized 32B)
      const uint4* p = (const uint4*)(Az + (long long)(m0 + arow) * K + k0 + ahalf);
      uint4 u0 = p[0], u1 = p[1];
      *(uint4*)(&As[arow * LDT + ahalf]) = u0;
      *(uint4*)(&As[arow * LDT + ahalf + 8]) = u1;
    }
    {   // B tile 32x64 -> LDS transposed Bt[n][k]
      const bf16* p = Bz + (long long)(k0 + bk) * N + n0 + bn0;
      bf16 tmp[16];
      *(uint4*)(tmp) = *(const uint4*)(p);
      *(uint4*)(tmp + 8) = *(const uint4*)(p + 8);
#pragma unroll
      for (int j = 0; j < 16; ++j) Bt[(bn0 + j) * LDT + bk] = tmp[j];
    }
    if (k0 + 32 < K) {   // prefetch next k-step's tiles (global_prefetch_b8)
      __builtin_prefetch(Az + (long long)(m0 + arow) * K + (k0 + 32) + ahalf, 0, 1);
      __builtin_prefetch(Bz + (long long)(k0 + 32 + bk) * N + n0 + bn0, 0, 1);
    }
    __syncthreads();
    v16bf af[2], bfr[2];
#pragma unroll
    for (int mi = 0; mi < 2; ++mi) af[mi] = frag_a(As, wr * 32 + mi * 16 + (lane & 15), lane);
#pragma unroll
    for (int ni = 0; ni < 2; ++ni) bfr[ni] = frag_b(Bt, wc * 32 + ni * 16 + (lane & 15), lane);
#pragma unroll
    for (int mi = 0; mi < 2; ++mi)
#pragma unroll
      for (int ni = 0; ni < 2; ++ni)
        acc[mi][ni] = __builtin_amdgcn_wmma_f32_16x16x32_bf16(
            false, af[mi], false, bfr[ni], (short)0, acc[mi][ni], false, false);
  }
  const int hi = lane >> 4, nn = lane & 15;
#pragma unroll
  for (int mi = 0; mi < 2; ++mi)
#pragma unroll
    for (int ni = 0; ni < 2; ++ni) {
      const int ncol = n0 + wc * 32 + ni * 16 + nn;
      const int mbase = m0 + wr * 32 + mi * 16 + hi * 8;
      const float bb = HAS_BIAS ? bias[ncol] : 0.0f;
      const long long base = (long long)mbase * N + ncol;
      const float* pr = HAS_RESID ? resid + (long long)z * sR + base : nullptr;
      float* pf = HAS_OUTF ? outF + (long long)z * sO + base : nullptr;
      bf16* pb = HAS_OUTB ? outB + (long long)z * sO + base : nullptr;
#pragma unroll
      for (int r = 0; r < 8; ++r) {
        float v = acc[mi][ni][r] + bb;
        if (HAS_RESID) v += pr[(long long)r * N];
        if (HAS_OUTF) pf[(long long)r * N] = v;
        if (HAS_OUTB) pb[(long long)r * N] = (bf16)v;
      }
    }
}

// ---------------------------------------------------------------------------
// Causal dilated conv as GEMM: Y[b,co,t] = sum_tap sum_ci W[tap,co,ci]*X[b,ci,t-(taps-1-tap)*dil]
// X: (B,Cin,S) bf16; W: (taps,Cout,Cin) bf16; bias per-row (Cout); +resid; ReLU.
// ---------------------------------------------------------------------------
template <bool HAS_RESID, bool HAS_OUTF, bool HAS_OUTB, bool RELU>
__global__ __launch_bounds__(128)
void k_conv(const bf16* __restrict__ X, const bf16* __restrict__ W,
            const float* __restrict__ bias, const float* resid,
            float* outF, bf16* outB,
            int Cin, int Cout, int taps, int dil) {
  __shared__ bf16 As[64 * LDT];
  __shared__ bf16 Bt[64 * LDT];
  const int tid = threadIdx.x, lane = tid & 31, wid = tid >> 5;
  const int wr = wid >> 1, wc = wid & 1;
  const int z = blockIdx.z;
  const int m0 = blockIdx.y * 64, n0 = blockIdx.x * 64;
  const bf16* Xz = X + (long long)z * Cin * S_;

  v8f acc[2][2] = {};
  const int arow = tid >> 1, ahalf = (tid & 1) * 16;
  const int bk = tid & 31, bn0 = (tid >> 5) * 16;

  for (int tap = 0; tap < taps; ++tap) {
    const int shift = (tap - (taps - 1)) * dil;       // causal: <= 0
    const bf16* Wt = W + (long long)tap * Cout * Cin;
    for (int k0 = 0; k0 < Cin; k0 += 32) {
      __syncthreads();
      {
        const uint4* p = (const uint4*)(Wt + (long long)(m0 + arow) * Cin + k0 + ahalf);
        uint4 u0 = p[0], u1 = p[1];
        *(uint4*)(&As[arow * LDT + ahalf]) = u0;
        *(uint4*)(&As[arow * LDT + ahalf + 8]) = u1;
      }
      {
        const bf16* px = Xz + (long long)(k0 + bk) * S_;
#pragma unroll
        for (int j = 0; j < 16; ++j) {
          int t = n0 + bn0 + j + shift;
          bf16 v = (bf16)0.0f;
          if (t >= 0 && t < S_) v = px[t];
          Bt[(bn0 + j) * LDT + bk] = v;
        }
      }
      if (k0 + 32 < Cin) {   // prefetch next k-step (global_prefetch_b8)
        __builtin_prefetch(Wt + (long long)(m0 + arow) * Cin + (k0 + 32) + ahalf, 0, 1);
        __builtin_prefetch(Xz + (long long)(k0 + 32 + bk) * S_ + n0 + bn0, 0, 1);
      }
      __syncthreads();
      v16bf af[2], bfr[2];
#pragma unroll
      for (int mi = 0; mi < 2; ++mi) af[mi] = frag_a(As, wr * 32 + mi * 16 + (lane & 15), lane);
#pragma unroll
      for (int ni = 0; ni < 2; ++ni) bfr[ni] = frag_b(Bt, wc * 32 + ni * 16 + (lane & 15), lane);
#pragma unroll
      for (int mi = 0; mi < 2; ++mi)
#pragma unroll
        for (int ni = 0; ni < 2; ++ni)
          acc[mi][ni] = __builtin_amdgcn_wmma_f32_16x16x32_bf16(
              false, af[mi], false, bfr[ni], (short)0, acc[mi][ni], false, false);
    }
  }
  const int hi = lane >> 4, nn = lane & 15;
#pragma unroll
  for (int mi = 0; mi < 2; ++mi)
#pragma unroll
    for (int ni = 0; ni < 2; ++ni) {
      const int ncol = n0 + wc * 32 + ni * 16 + nn;
      const int mbase = m0 + wr * 32 + mi * 16 + hi * 8;
      const long long base = ((long long)z * Cout + mbase) * S_ + ncol;
      const float* pr = HAS_RESID ? resid + base : nullptr;
      float* pf = HAS_OUTF ? outF + base : nullptr;
      bf16* pb = HAS_OUTB ? outB + base : nullptr;
#pragma unroll
      for (int r = 0; r < 8; ++r) {
        float v = acc[mi][ni][r] + bias[mbase + r];
        if (HAS_RESID) v += pr[(long long)r * S_];
        if (RELU) v = fmaxf(v, 0.0f);
        if (HAS_OUTF) pf[(long long)r * S_] = v;
        if (HAS_OUTB) pb[(long long)r * S_] = (bf16)v;
      }
    }
}

// ---------------------------------------------------------------------------
// Elementwise / reduction kernels
// ---------------------------------------------------------------------------
__global__ __launch_bounds__(256) void k_cvt(const float* s, bf16* d, int n) {
  int i = blockIdx.x * 256 + threadIdx.x;
  if (i < n) d[i] = (bf16)s[i];
}
// (Cout,Cin,T) f32 -> (T,Cout,Cin) bf16
__global__ __launch_bounds__(256) void k_cvtw(const float* s, bf16* d, int Cout, int Cin, int T) {
  int n = Cout * Cin * T;
  int i = blockIdx.x * 256 + threadIdx.x;
  if (i >= n) return;
  int ci = i % Cin, co = (i / Cin) % Cout, t = i / (Cin * Cout);
  d[i] = (bf16)s[((long long)co * Cin + ci) * T + t];
}
// (B,S,D) f32 -> (B,D,S) bf16
__global__ __launch_bounds__(256) void k_tr_sd2ds(const float* s, bf16* d) {
  long long n = (long long)B_ * S_ * D_;
  long long i = (long long)blockIdx.x * 256 + threadIdx.x;
  if (i >= n) return;
  int ss = (int)(i % S_); long long r = i / S_;
  int dd = (int)(r % D_); int b = (int)(r / D_);
  d[i] = (bf16)s[((long long)b * S_ + ss) * D_ + dd];
}
// (B,C,S) f32 -> (B,S,C) bf16
__global__ __launch_bounds__(256) void k_tr_ds2sd(const float* s, bf16* d, int C) {
  long long n = (long long)B_ * S_ * C;
  long long i = (long long)blockIdx.x * 256 + threadIdx.x;
  if (i >= n) return;
  int c = (int)(i % C); long long r = i / C;
  int ss = (int)(r % S_); int b = (int)(r / S_);
  d[i] = (bf16)s[((long long)b * C + c) * S_ + ss];
}
// h += sinusoidal PE; also emit bf16 copy
__global__ __launch_bounds__(256) void k_pe(float* h, bf16* hb) {
  long long n = (long long)B_ * S_ * D_;
  long long i = (long long)blockIdx.x * 256 + threadIdx.x;
  if (i >= n) return;
  int d = (int)(i % D_); int ss = (int)((i / D_) % S_);
  float freq = expf(-(float)(d & ~1) * (9.210340371976184f / (float)D_));
  float ang = (float)ss * freq;
  float v = h[i] + ((d & 1) ? cosf(ang) : sinf(ang));
  h[i] = v; hb[i] = (bf16)v;
}
// q (B,S,D) f32 -> elu(q)+1 as (B,H,S,HD) bf16
__global__ __launch_bounds__(256) void k_elu_q(const float* q, bf16* qf) {
  long long n = (long long)B_ * S_ * D_;
  long long i = (long long)blockIdx.x * 256 + threadIdx.x;
  if (i >= n) return;
  int d = (int)(i % D_); int ss = (int)((i / D_) % S_); int b = (int)(i / ((long long)D_ * S_));
  int hh = d >> 6, dd = d & 63;
  float v = q[i];
  float e = v > 0.f ? v : (expf(v) - 1.f);
  qf[(((long long)(b * H_ + hh) * S_ + ss) * HD_) + dd] = (bf16)(e + 1.f);
}
// k (B,S,D) f32 -> elu(k)+1 transposed as (B,H,HD,S) bf16
__global__ __launch_bounds__(256) void k_elu_kT(const float* kk, bf16* kfT) {
  long long n = (long long)B_ * S_ * D_;
  long long i = (long long)blockIdx.x * 256 + threadIdx.x;
  if (i >= n) return;
  int d = (int)(i % D_); int ss = (int)((i / D_) % S_); int b = (int)(i / ((long long)D_ * S_));
  int hh = d >> 6, dd = d & 63;
  float v = kk[i];
  float e = v > 0.f ? v : (expf(v) - 1.f);
  kfT[(((long long)(b * H_ + hh) * HD_ + dd) * S_) + ss] = (bf16)(e + 1.f);
}
// v (B,S,D) f32 -> (B,H,S,HD) bf16
__global__ __launch_bounds__(256) void k_vshape(const float* v, bf16* vb) {
  long long n = (long long)B_ * S_ * D_;
  long long i = (long long)blockIdx.x * 256 + threadIdx.x;
  if (i >= n) return;
  int d = (int)(i % D_); int ss = (int)((i / D_) % S_); int b = (int)(i / ((long long)D_ * S_));
  int hh = d >> 6, dd = d & 63;
  vb[(((long long)(b * H_ + hh) * S_ + ss) * HD_) + dd] = (bf16)v[i];
}
// row sums of kfT (rows = B*H*HD, length S), one wave per row
__global__ __launch_bounds__(256) void k_ksum(const bf16* kfT, float* ksum, int rows) {
  int wid = threadIdx.x >> 5, lane = threadIdx.x & 31;
  int row = blockIdx.x * 8 + wid;
  if (row >= rows) return;
  const bf16* p = kfT + (long long)row * S_;
  float s = 0.f;
  for (int i = lane; i < S_; i += 32) s += (float)p[i];
#pragma unroll
  for (int m = 16; m > 0; m >>= 1) s += __shfl_xor(s, m, 32);
  if (lane == 0) ksum[row] = s;
}
// norm[b,h,s] = max(qf . ksum, 1e-6)
__global__ __launch_bounds__(256) void k_norm(const bf16* qf, const float* ksum, float* nrm) {
  int n = B_ * H_ * S_;
  int i = blockIdx.x * 256 + threadIdx.x;
  if (i >= n) return;
  int ss = i % S_, bh = i / S_;
  const bf16* p = qf + ((long long)bh * S_ + ss) * HD_;
  const float* ks = ksum + bh * HD_;
  float a = 0.f;
#pragma unroll
  for (int d = 0; d < HD_; ++d) a += (float)p[d] * ks[d];
  nrm[i] = fmaxf(a, 1e-6f);
}
// attnraw (B,H,S,HD) / norm -> (B,S,D) bf16
__global__ __launch_bounds__(256) void k_merge(const float* raw, const float* nrm, bf16* out) {
  long long n = (long long)B_ * H_ * S_ * HD_;
  long long i = (long long)blockIdx.x * 256 + threadIdx.x;
  if (i >= n) return;
  int d = (int)(i & 63);
  int ss = (int)((i >> 6) % S_);
  int bh = (int)(i / ((long long)S_ * HD_));
  int b = bh >> 3, hh = bh & 7;
  out[((long long)b * S_ + ss) * D_ + hh * HD_ + d] = (bf16)(raw[i] / nrm[(long long)bh * S_ + ss]);
}
// LayerNorm over D=512, one block per row; writes f32 + bf16
__global__ __launch_bounds__(256) void k_ln(const float* y, const float* g, const float* be,
                                            float* ho, bf16* hb) {
  __shared__ float red[256];
  int row = blockIdx.x, tid = threadIdx.x;
  const float* p = y + (long long)row * D_;
  float v0 = p[tid], v1 = p[tid + 256];
  red[tid] = v0 + v1; __syncthreads();
  for (int s = 128; s > 0; s >>= 1) { if (tid < s) red[tid] += red[tid + s]; __syncthreads(); }
  float mu = red[0] * (1.f / D_);
  __syncthreads();
  float d0 = v0 - mu, d1 = v1 - mu;
  red[tid] = d0 * d0 + d1 * d1; __syncthreads();
  for (int s = 128; s > 0; s >>= 1) { if (tid < s) red[tid] += red[tid + s]; __syncthreads(); }
  float rstd = rsqrtf(red[0] * (1.f / D_) + 1e-5f);
  float o0 = d0 * rstd * g[tid] + be[tid];
  float o1 = d1 * rstd * g[tid + 256] + be[tid + 256];
  long long base = (long long)row * D_;
  ho[base + tid] = o0; ho[base + tid + 256] = o1;
  hb[base + tid] = (bf16)o0; hb[base + tid + 256] = (bf16)o1;
}
// mean over S then dot with W_out; one block per batch
__global__ __launch_bounds__(256) void k_pool(const float* h, const float* wout,
                                              const float* bout, float* out) {
  __shared__ float red[256];
  int b = blockIdx.x, tid = threadIdx.x;
  const float* p = h + (long long)b * S_ * D_;
  float a0 = 0.f, a1 = 0.f;
  for (int s = 0; s < S_; ++s) {
    a0 += p[(long long)s * D_ + tid];
    a1 += p[(long long)s * D_ + tid + 256];
  }
  red[tid] = (a0 * wout[tid] + a1 * wout[tid + 256]) * (1.f / S_);
  __syncthreads();
  for (int s = 128; s > 0; s >>= 1) { if (tid < s) red[tid] += red[tid + s]; __syncthreads(); }
  if (tid == 0) out[b] = red[0] + bout[0];
}

// ---------------------------------------------------------------------------
// Host orchestration
// ---------------------------------------------------------------------------
static inline int ceil_div(long long a, int b) { return (int)((a + b - 1) / b); }

extern "C" void kernel_launch(void* const* d_in, const int* in_sizes, int n_in,
                              void* d_out, int out_size, void* d_ws, size_t ws_size,
                              hipStream_t stream) {
  (void)in_sizes; (void)n_in; (void)out_size; (void)ws_size;
  // inputs (setup_inputs dict order)
  const float* x     = (const float*)d_in[0];
  const float* W_in  = (const float*)d_in[1];
  const float* b_in  = (const float*)d_in[2];
  const float* c1w0  = (const float*)d_in[3];
  const float* c1b0  = (const float*)d_in[4];
  const float* c2w0  = (const float*)d_in[5];
  const float* c2b0  = (const float*)d_in[6];
  const float* dsw0  = (const float*)d_in[7];
  const float* dsb0  = (const float*)d_in[8];
  const float* c1w1  = (const float*)d_in[9];
  const float* c1b1  = (const float*)d_in[10];
  const float* c2w1  = (const float*)d_in[11];
  const float* c2b1  = (const float*)d_in[12];
  const float* c1w2  = (const float*)d_in[13];
  const float* c1b2  = (const float*)d_in[14];
  const float* c2w2  = (const float*)d_in[15];
  const float* c2b2  = (const float*)d_in[16];
  const float* dsw2  = (const float*)d_in[17];
  const float* dsb2  = (const float*)d_in[18];
  const float* W_tcn = (const float*)d_in[19];
  const float* b_tcn = (const float*)d_in[20];
  const float* Wq    = (const float*)d_in[21];
  const float* Wk    = (const float*)d_in[22];
  const float* Wv    = (const float*)d_in[23];
  const float* Wo    = (const float*)d_in[24];
  const float* bq    = (const float*)d_in[25];
  const float* bk    = (const float*)d_in[26];
  const float* bv    = (const float*)d_in[27];
  const float* bo    = (const float*)d_in[28];
  const float* ln_b  = (const float*)d_in[29];
  const float* ln_g  = (const float*)d_in[30];
  const float* W_out = (const float*)d_in[31];
  const float* b_out = (const float*)d_in[32];
  float* out = (float*)d_out;

  const long long BS  = (long long)B_ * S_;          // 32768
  const long long BSD = BS * D_;                     // 16.7M elems

  // workspace bump allocator
  char* ws = (char*)d_ws; size_t off = 0;
  auto alloc = [&](size_t bytes) -> char* {
    char* p = ws + off; off += (bytes + 255) & ~(size_t)255; return p;
  };
  float* h   = (float*)alloc(BSD * 4);      // persistent activations f32
  float* f1  = (float*)alloc(BSD * 4);      // r / attnraw
  float* f2  = (float*)alloc(BSD * 4);      // tA / q
  float* f3  = (float*)alloc(BSD * 4);      // tB / k
  float* f4  = (float*)alloc(BSD * 4);      // v / y
  bf16* hb   = (bf16*)alloc(BSD * 2);
  bf16* b1   = (bf16*)alloc(BSD * 2);       // t0b / qf
  bf16* b2   = (bf16*)alloc(BSD * 2);       // o1b / kfT
  bf16* b3   = (bf16*)alloc(BSD * 2);       // tAb / vb
  bf16* b4   = (bf16*)alloc(BSD * 2);       // tBb / tcn_bsd / attn_bsd
  bf16* xb   = (bf16*)alloc(BS * IN_ * 2);
  bf16* winb = (bf16*)alloc(IN_ * D_ * 2);
  bf16* wtcb = (bf16*)alloc(D_ * D_ * 2);
  bf16* wqb  = (bf16*)alloc((size_t)NL_ * D_ * D_ * 2);
  bf16* wkb  = (bf16*)alloc((size_t)NL_ * D_ * D_ * 2);
  bf16* wvb  = (bf16*)alloc((size_t)NL_ * D_ * D_ * 2);
  bf16* wob  = (bf16*)alloc((size_t)NL_ * D_ * D_ * 2);
  bf16* c1w0b = (bf16*)alloc((size_t)3 * 256 * 512 * 2);
  bf16* c2w0b = (bf16*)alloc((size_t)3 * 256 * 256 * 2);
  bf16* ds0b  = (bf16*)alloc((size_t)256 * 512 * 2);
  bf16* c1w1b = (bf16*)alloc((size_t)3 * 256 * 256 * 2);
  bf16* c2w1b = (bf16*)alloc((size_t)3 * 256 * 256 * 2);
  bf16* c1w2b = (bf16*)alloc((size_t)3 * 512 * 256 * 2);
  bf16* c2w2b = (bf16*)alloc((size_t)3 * 512 * 512 * 2);
  bf16* ds2b  = (bf16*)alloc((size_t)512 * 256 * 2);
  float* ksumF = (float*)alloc((size_t)B_ * H_ * HD_ * 4);
  float* kvF   = (float*)alloc((size_t)B_ * H_ * HD_ * HD_ * 4);
  bf16*  kvB   = (bf16*)alloc((size_t)B_ * H_ * HD_ * HD_ * 2);
  float* nrmF  = (float*)alloc((size_t)B_ * H_ * S_ * 4);

  // ---- weight / input conversions to bf16 ----
  k_cvt<<<ceil_div(BS * IN_, 256), 256, 0, stream>>>(x, xb, (int)(BS * IN_));
  k_cvt<<<ceil_div(IN_ * D_, 256), 256, 0, stream>>>(W_in, winb, IN_ * D_);
  k_cvt<<<ceil_div(D_ * D_, 256), 256, 0, stream>>>(W_tcn, wtcb, D_ * D_);
  k_cvt<<<ceil_div(NL_ * D_ * D_, 256), 256, 0, stream>>>(Wq, wqb, NL_ * D_ * D_);
  k_cvt<<<ceil_div(NL_ * D_ * D_, 256), 256, 0, stream>>>(Wk, wkb, NL_ * D_ * D_);
  k_cvt<<<ceil_div(NL_ * D_ * D_, 256), 256, 0, stream>>>(Wv, wvb, NL_ * D_ * D_);
  k_cvt<<<ceil_div(NL_ * D_ * D_, 256), 256, 0, stream>>>(Wo, wob, NL_ * D_ * D_);
  k_cvtw<<<ceil_div(3 * 256 * 512, 256), 256, 0, stream>>>(c1w0, c1w0b, 256, 512, 3);
  k_cvtw<<<ceil_div(3 * 256 * 256, 256), 256, 0, stream>>>(c2w0, c2w0b, 256, 256, 3);
  k_cvtw<<<ceil_div(256 * 512, 256), 256, 0, stream>>>(dsw0, ds0b, 256, 512, 1);
  k_cvtw<<<ceil_div(3 * 256 * 256, 256), 256, 0, stream>>>(c1w1, c1w1b, 256, 256, 3);
  k_cvtw<<<ceil_div(3 * 256 * 256, 256), 256, 0, stream>>>(c2w1, c2w1b, 256, 256, 3);
  k_cvtw<<<ceil_div(3 * 512 * 256, 256), 256, 0, stream>>>(c1w2, c1w2b, 512, 256, 3);
  k_cvtw<<<ceil_div(3 * 512 * 512, 256), 256, 0, stream>>>(c2w2, c2w2b, 512, 512, 3);
  k_cvtw<<<ceil_div(512 * 256, 256), 256, 0, stream>>>(dsw2, ds2b, 512, 256, 1);

  const int eb = ceil_div(BSD, 256);        // elementwise blocks over BSD
  const dim3 gproj(D_ / 64, (int)(BS / 64), 1);

  // ---- input projection: h = x @ W_in + b_in ----
  k_gemm<true, false, true, false><<<gproj, 128, 0, stream>>>(
      xb, winb, b_in, nullptr, h, nullptr, (int)BS, D_, IN_, 0, 0, 0, 0);
  // (B,S,D) -> (B,D,S) bf16 for TCN
  k_tr_sd2ds<<<eb, 256, 0, stream>>>(h, b1);

  // ---- TCN block 0 (512->256, dil 1, with downsample) ----
  dim3 g256(S_ / 64, 256 / 64, B_), g512(S_ / 64, 512 / 64, B_);
  k_conv<false, false, true, true><<<g256, 128, 0, stream>>>(
      b1, c1w0b, c1b0, nullptr, nullptr, b2, 512, 256, 3, 1);
  k_conv<false, true, false, false><<<g256, 128, 0, stream>>>(
      b1, ds0b, dsb0, nullptr, f1, nullptr, 512, 256, 1, 1);
  k_conv<true, true, true, true><<<g256, 128, 0, stream>>>(
      b2, c2w0b, c2b0, f1, f3, b4, 256, 256, 3, 1);
  // ---- TCN block 1 (256->256, dil 2) ----
  k_conv<false, false, true, true><<<g256, 128, 0, stream>>>(
      b4, c1w1b, c1b1, nullptr, nullptr, b2, 256, 256, 3, 2);
  k_conv<true, true, true, true><<<g256, 128, 0, stream>>>(
      b2, c2w1b, c2b1, f3, f2, b3, 256, 256, 3, 2);
  // ---- TCN block 2 (256->512, dil 4, with downsample) ----
  k_conv<false, false, true, true><<<g512, 128, 0, stream>>>(
      b3, c1w2b, c1b2, nullptr, nullptr, b2, 256, 512, 3, 4);
  k_conv<false, true, false, false><<<g512, 128, 0, stream>>>(
      b3, ds2b, dsb2, nullptr, f1, nullptr, 256, 512, 1, 4);
  k_conv<true, true, false, true><<<g512, 128, 0, stream>>>(
      b2, c2w2b, c2b2, f1, f3, nullptr, 512, 512, 3, 4);

  // ---- h = h + tcn^T @ W_tcn + b_tcn, then + positional encoding ----
  k_tr_ds2sd<<<eb, 256, 0, stream>>>(f3, b4, D_);
  k_gemm<true, true, true, false><<<gproj, 128, 0, stream>>>(
      b4, wtcb, b_tcn, h, h, nullptr, (int)BS, D_, D_, 0, 0, 0, 0);
  k_pe<<<eb, 256, 0, stream>>>(h, hb);

  // ---- linear-attention layers ----
  for (int l = 0; l < NL_; ++l) {
    const long long w_off = (long long)l * D_ * D_;
    // Q/K/V projections
    k_gemm<true, false, true, false><<<gproj, 128, 0, stream>>>(
        hb, wqb + w_off, bq + l * D_, nullptr, f2, nullptr, (int)BS, D_, D_, 0, 0, 0, 0);
    k_gemm<true, false, true, false><<<gproj, 128, 0, stream>>>(
        hb, wkb + w_off, bk + l * D_, nullptr, f3, nullptr, (int)BS, D_, D_, 0, 0, 0, 0);
    k_gemm<true, false, true, false><<<gproj, 128, 0, stream>>>(
        hb, wvb + w_off, bv + l * D_, nullptr, f4, nullptr, (int)BS, D_, D_, 0, 0, 0, 0);
    // feature maps + head reshapes
    k_elu_q<<<eb, 256, 0, stream>>>(f2, b1);       // qf  (B,H,S,HD)
    k_elu_kT<<<eb, 256, 0, stream>>>(f3, b2);      // kfT (B,H,HD,S)
    k_vshape<<<eb, 256, 0, stream>>>(f4, b3);      // v   (B,H,S,HD)
    // ksum over sequence
    k_ksum<<<(B_ * H_ * HD_) / 8, 256, 0, stream>>>(b2, ksumF, B_ * H_ * HD_);
    // kv = kf^T @ v : batched (B*H), 64x64, K=4096
    k_gemm<false, false, true, true><<<dim3(1, 1, B_ * H_), 128, 0, stream>>>(
        b2, b3, nullptr, nullptr, kvF, kvB, HD_, HD_, S_,
        (long long)HD_ * S_, (long long)S_ * HD_, 0, (long long)HD_ * HD_);
    // out = qf @ kv : batched (B*H), 4096x64, K=64
    k_gemm<false, false, true, false><<<dim3(1, S_ / 64, B_ * H_), 128, 0, stream>>>(
        b1, kvB, nullptr, nullptr, f1, nullptr, S_, HD_, HD_,
        (long long)S_ * HD_, (long long)HD_ * HD_, 0, (long long)S_ * HD_);
    // normalization + head merge
    k_norm<<<ceil_div((long long)B_ * H_ * S_, 256), 256, 0, stream>>>(b1, ksumF, nrmF);
    k_merge<<<eb, 256, 0, stream>>>(f1, nrmF, b4); // (B,S,D) bf16
    // output projection + residual, then LayerNorm -> h (f32 + bf16)
    k_gemm<true, true, true, false><<<gproj, 128, 0, stream>>>(
        b4, wob + w_off, bo + l * D_, h, f4, nullptr, (int)BS, D_, D_, 0, 0, 0, 0);
    k_ln<<<(int)BS, 256, 0, stream>>>(f4, ln_g + l * D_, ln_b + l * D_, h, hb);
  }

  // ---- mean-pool over S and final projection ----
  k_pool<<<B_, 256, 0, stream>>>(h, W_out, b_out, out);
}
